// MultiHeadAttention_33749853012290
// MI455X (gfx1250) — compile-verified, matrix operand order checked by reading, not run on hardware
//
#include <hip/hip_runtime.h>
#include <hip/hip_bf16.h>

// ---------------------------------------------------------------------------
// MHA forward for MI455X (gfx1250, wave32, WMMA).
// All four matmul stages use v_wmma_f32_16x16x32_bf16 with register blocking;
// fp32->bf16 conversion is hoisted into a one-shot pre-pass.
// ---------------------------------------------------------------------------

typedef __bf16 bf16;
typedef __attribute__((ext_vector_type(16))) __bf16 v16bf;
typedef __attribute__((ext_vector_type(8)))  __bf16 v8bf;
typedef __attribute__((ext_vector_type(8)))  float  v8f;
typedef __attribute__((ext_vector_type(4)))  float  v4f;

constexpr int Bc = 2;
constexpr int Tc = 2048;
constexpr int Ec = 1024;
constexpr int Hc = 16;
constexpr int Dc = 64;   // Ec / Hc

// D = A(16x32 bf16) * B(32x16 bf16, row-of-K per lane) + C(16x16 f32)
__device__ __forceinline__ v8f wmma_bf16(v16bf a, v16bf b, v8f c) {
  return __builtin_amdgcn_wmma_f32_16x16x32_bf16(
      /*neg_a=*/false, a, /*neg_b=*/false, b,
      /*c_mod=*/(short)0, c, /*reuse_a=*/false, /*reuse_b=*/false);
}

// 16x32 bf16 fragment from row-major memory, row stride `ld` elements.
// ISA 16-bit A layout (wave32): lane L holds row (L&15); K-chunks of 8 at
// columns (L>>4)*8 and 16+(L>>4)*8.  Two 16B loads per lane.
__device__ __forceinline__ v16bf load_frag_bf16(const bf16* base, int ld) {
  const int lane = threadIdx.x & 31;
  const bf16* p = base + (size_t)(lane & 15) * ld + (lane >> 4) * 8;
  v8bf lo = *(const v8bf*)(p);
  v8bf hi = *(const v8bf*)(p + 16);
  return __builtin_shufflevector(lo, hi, 0, 1, 2, 3, 4, 5, 6, 7,
                                 8, 9, 10, 11, 12, 13, 14, 15);
}

// ---------------------------------------------------------------------------
// Kernel 0: one-shot fp32 -> bf16 conversion (8 elements / thread).
// ---------------------------------------------------------------------------
__global__ __launch_bounds__(256) void cvt_f32_bf16_kernel(
    const float* __restrict__ src, bf16* __restrict__ dst) {
  const size_t i = ((size_t)blockIdx.x * 256 + threadIdx.x) * 8;
  v4f a = *(const v4f*)(src + i);
  v4f b = *(const v4f*)(src + i + 4);
  v8bf o;
#pragma unroll
  for (int j = 0; j < 4; ++j) {
    o[j]     = (bf16)a[j];
    o[4 + j] = (bf16)b[j];
  }
  *(v8bf*)(dst + i) = o;
}

// ---------------------------------------------------------------------------
// Kernel 1: fused QKV projection, 32x64 output tile per wave (2x4 WMMA tiles).
// z = 0/1/2 selects Q/K/V.  Q,K -> bf16 [B,H,T,D];  V -> bf16 transposed
// [B,H,D,T] so the later P@V WMMA B-operand is a plain row-major load.
// ---------------------------------------------------------------------------
__global__ __launch_bounds__(256) void qkv_proj_kernel(
    const bf16* __restrict__ xb,
    const bf16* __restrict__ Wqb, const bf16* __restrict__ Wkb,
    const bf16* __restrict__ Wvb,
    const float* __restrict__ bq, const float* __restrict__ bk,
    const float* __restrict__ bv,
    bf16* __restrict__ Qws, bf16* __restrict__ Kws, bf16* __restrict__ Vtws) {
  const int wave = threadIdx.x >> 5;
  const int lane = threadIdx.x & 31;
  const int m0 = (blockIdx.x * 8 + wave) * 32;  // token-row block
  const int n0 = blockIdx.y * 64;               // feature-col block
  const int which = blockIdx.z;                 // 0:Q 1:K 2:V

  const bf16* W     = (which == 0) ? Wqb : (which == 1) ? Wkb : Wvb;
  const float* bias = (which == 0) ? bq : (which == 1) ? bk : bv;

  v8f acc[2][4] = {};
  for (int k = 0; k < Ec; k += 32) {
    const v16bf a0 = load_frag_bf16(xb + (size_t)m0 * Ec + k, Ec);
    const v16bf a1 = load_frag_bf16(xb + (size_t)(m0 + 16) * Ec + k, Ec);
#pragma unroll
    for (int j = 0; j < 4; ++j) {
      const v16bf b = load_frag_bf16(W + (size_t)(n0 + 16 * j) * Ec + k, Ec);
      acc[0][j] = wmma_bf16(a0, b, acc[0][j]);
      acc[1][j] = wmma_bf16(a1, b, acc[1][j]);
    }
  }

  const int hi = lane >> 4, lo = lane & 15;
#pragma unroll
  for (int j = 0; j < 4; ++j) {
    const int n = n0 + 16 * j + lo;
    const float bn = bias[n];
    const int h = n >> 6;        // n / D
    const int d = n & (Dc - 1);  // n % D
#pragma unroll
    for (int i = 0; i < 2; ++i) {
      const int mbase = m0 + 16 * i + 8 * hi;   // 8 consecutive token rows
      const int b_ = mbase >> 11;               // batch (T = 2048)
      const int t0 = mbase & (Tc - 1);
      const size_t bh = (size_t)b_ * Hc + h;
      if (which == 2) {
        // V transposed: fixed d, 8 consecutive t -> one 16B store
        v8bf o;
#pragma unroll
        for (int r = 0; r < 8; ++r) o[r] = (bf16)(acc[i][j][r] + bn);
        *(v8bf*)(Vtws + (bh * Dc + d) * Tc + t0) = o;
      } else {
        bf16* dst = ((which == 0) ? Qws : Kws) + (bh * Tc + t0) * Dc + d;
#pragma unroll
        for (int r = 0; r < 8; ++r) dst[(size_t)r * Dc] = (bf16)(acc[i][j][r] + bn);
      }
    }
  }
}

// ---------------------------------------------------------------------------
// Kernel 2: flash attention.  One wave owns a 32-row q-tile (2 WMMA q-subtiles
// sharing every K/V fragment); loop keys in chunks of 32 with online softmax.
// 16 WMMAs per key-chunk.  C layout: element (M = r + 8*(lane>=16), N=lane&15).
// ---------------------------------------------------------------------------
__global__ __launch_bounds__(256) void flash_attn_kernel(
    const bf16* __restrict__ Qws, const bf16* __restrict__ Kws,
    const bf16* __restrict__ Vtws, bf16* __restrict__ attnws) {
  __shared__ bf16 pshared[8][2][16 * 32];  // per-wave P staging (C -> A layout)

  const int wave = threadIdx.x >> 5;
  const int lane = threadIdx.x & 31;
  const int hi = lane >> 4, lo = lane & 15;
  const int b = blockIdx.z, h = blockIdx.y;
  const int q0 = (blockIdx.x * 8 + wave) * 32;

  const bf16* Qbase = Qws + (((size_t)b * Hc + h) * Tc + q0) * Dc;
  const bf16* Kbase = Kws + ((size_t)b * Hc + h) * Tc * Dc;
  const bf16* Vbase = Vtws + ((size_t)b * Hc + h) * Dc * Tc;

  // Q tiles (2 x 16 x 64) in registers: [qsub][k-chunk of 32]
  v16bf qa[2][2];
#pragma unroll
  for (int qs = 0; qs < 2; ++qs) {
    qa[qs][0] = load_frag_bf16(Qbase + (size_t)qs * 16 * Dc + 0, Dc);
    qa[qs][1] = load_frag_bf16(Qbase + (size_t)qs * 16 * Dc + 32, Dc);
  }

  v8f acc[2][4] = {};
  float m_i[2][8], l_i[2][8];
#pragma unroll
  for (int qs = 0; qs < 2; ++qs)
#pragma unroll
    for (int r = 0; r < 8; ++r) { m_i[qs][r] = -1e30f; l_i[qs][r] = 0.0f; }
  const float scale = 0.125f;  // 1/sqrt(D)

  for (int kc = 0; kc < Tc; kc += 32) {
    const bf16* kb = Kbase + (size_t)kc * Dc;
    const bf16* vb = Vbase + kc;

    // Prefetch next key chunk's K rows and V^T rows (global_prefetch_b8).
    if (kc + 32 < Tc) {
      __builtin_prefetch((const void*)(kb + 32 * Dc + (size_t)lane * Dc), 0, 1);
      __builtin_prefetch((const void*)(vb + 32 + (size_t)(2 * lane) * Tc), 0, 1);
    }

    // K fragments shared by both q-subtiles: [key-half][d-chunk]
    v16bf kf[2][2];
    kf[0][0] = load_frag_bf16(kb, Dc);
    kf[0][1] = load_frag_bf16(kb + 32, Dc);
    kf[1][0] = load_frag_bf16(kb + 16 * Dc, Dc);
    kf[1][1] = load_frag_bf16(kb + 16 * Dc + 32, Dc);

#pragma unroll
    for (int qs = 0; qs < 2; ++qs) {
      v8f s0 = {}, s1 = {};
      s0 = wmma_bf16(qa[qs][0], kf[0][0], s0);
      s0 = wmma_bf16(qa[qs][1], kf[0][1], s0);
      s1 = wmma_bf16(qa[qs][0], kf[1][0], s1);
      s1 = wmma_bf16(qa[qs][1], kf[1][1], s1);

      bf16* pbuf = &pshared[wave][qs][0];
#pragma unroll
      for (int r = 0; r < 8; ++r) {
        const float e0 = s0[r] * scale;
        const float e1 = s1[r] * scale;
        float mx = fmaxf(e0, e1);
#pragma unroll
        for (int off = 8; off >= 1; off >>= 1)
          mx = fmaxf(mx, __shfl_xor(mx, off, 32));
        const float mnew  = fmaxf(m_i[qs][r], mx);
        const float alpha = __expf(m_i[qs][r] - mnew);
        const float p0 = __expf(e0 - mnew);
        const float p1 = __expf(e1 - mnew);
        float rs = p0 + p1;
#pragma unroll
        for (int off = 8; off >= 1; off >>= 1)
          rs += __shfl_xor(rs, off, 32);
        l_i[qs][r] = l_i[qs][r] * alpha + rs;
        m_i[qs][r] = mnew;
#pragma unroll
        for (int j = 0; j < 4; ++j) acc[qs][j][r] *= alpha;
        const int row = r + 8 * hi;
        pbuf[row * 32 + lo]      = (bf16)p0;
        pbuf[row * 32 + 16 + lo] = (bf16)p1;
      }
    }
    __syncthreads();  // make LDS P visible for the A-layout reload

    // O(32x64) += P(32x32) . V^T rows (d, keys kc..kc+31)
    v16bf vf[4];
#pragma unroll
    for (int j = 0; j < 4; ++j)
      vf[j] = load_frag_bf16(vb + (size_t)j * 16 * Tc, Tc);
#pragma unroll
    for (int qs = 0; qs < 2; ++qs) {
      const v16bf pa = load_frag_bf16(&pshared[wave][qs][0], 32);
#pragma unroll
      for (int j = 0; j < 4; ++j)
        acc[qs][j] = wmma_bf16(pa, vf[j], acc[qs][j]);
    }
    __syncthreads();  // P tiles consumed; safe to overwrite next iteration
  }

  // ---- normalize, write bf16 attn output in [B, T, E] (e = h*D + d)
#pragma unroll
  for (int qs = 0; qs < 2; ++qs)
#pragma unroll
    for (int r = 0; r < 8; ++r) {
      const float inv = 1.0f / l_i[qs][r];
      const int t = q0 + 16 * qs + r + 8 * hi;
      const size_t rowoff = ((size_t)b * Tc + t) * Ec + (size_t)h * Dc;
#pragma unroll
      for (int j = 0; j < 4; ++j)
        attnws[rowoff + j * 16 + lo] = (bf16)(acc[qs][j][r] * inv);
    }
}

// ---------------------------------------------------------------------------
// Kernel 3: output projection, 32x64 tile per wave.  out = attn @ Wo^T + bo.
// ---------------------------------------------------------------------------
__global__ __launch_bounds__(256) void out_proj_kernel(
    const bf16* __restrict__ attnws, const bf16* __restrict__ Wob,
    const float* __restrict__ bo, float* __restrict__ out) {
  const int wave = threadIdx.x >> 5;
  const int lane = threadIdx.x & 31;
  const int m0 = (blockIdx.x * 8 + wave) * 32;
  const int n0 = blockIdx.y * 64;

  v8f acc[2][4] = {};
  for (int k = 0; k < Ec; k += 32) {
    const v16bf a0 = load_frag_bf16(attnws + (size_t)m0 * Ec + k, Ec);
    const v16bf a1 = load_frag_bf16(attnws + (size_t)(m0 + 16) * Ec + k, Ec);
#pragma unroll
    for (int j = 0; j < 4; ++j) {
      const v16bf b = load_frag_bf16(Wob + (size_t)(n0 + 16 * j) * Ec + k, Ec);
      acc[0][j] = wmma_bf16(a0, b, acc[0][j]);
      acc[1][j] = wmma_bf16(a1, b, acc[1][j]);
    }
  }

  const int hi = lane >> 4, lo = lane & 15;
#pragma unroll
  for (int j = 0; j < 4; ++j) {
    const float bn = bo[n0 + 16 * j + lo];
#pragma unroll
    for (int i = 0; i < 2; ++i) {
      float* dst = out + (size_t)(m0 + 16 * i + 8 * hi) * Ec + n0 + 16 * j + lo;
#pragma unroll
      for (int r = 0; r < 8; ++r) dst[(size_t)r * Ec] = acc[i][j][r] + bn;
    }
  }
}

// ---------------------------------------------------------------------------
extern "C" void kernel_launch(void* const* d_in, const int* in_sizes, int n_in,
                              void* d_out, int out_size, void* d_ws,
                              size_t ws_size, hipStream_t stream) {
  const float* x  = (const float*)d_in[0];
  const float* Wq = (const float*)d_in[1];
  const float* bq = (const float*)d_in[2];
  const float* Wk = (const float*)d_in[3];
  const float* bk = (const float*)d_in[4];
  const float* Wv = (const float*)d_in[5];
  const float* bv = (const float*)d_in[6];
  const float* Wo = (const float*)d_in[7];
  const float* bo = (const float*)d_in[8];
  float* out = (float*)d_out;

  // Workspace layout (bf16 elements):
  //   xb (4M) | Wq (1M) | Wk (1M) | Wv (1M) | Wo (1M) |
  //   Q (4M)  | K (4M)  | V^T (4M) | attn (4M)          = 24M elems = 48MB
  const size_t nX = (size_t)Bc * Tc * Ec;  // 4,194,304
  const size_t nW = (size_t)Ec * Ec;       // 1,048,576
  bf16* xb   = (bf16*)d_ws;
  bf16* Wqb  = xb + nX;
  bf16* Wkb  = Wqb + nW;
  bf16* Wvb  = Wkb + nW;
  bf16* Wob  = Wvb + nW;
  bf16* Qws  = Wob + nW;
  bf16* Kws  = Qws + nX;
  bf16* Vtws = Kws + nX;
  bf16* attn = Vtws + nX;

  // 0) fp32 -> bf16 pre-pass (removes all conversion from GEMM hot loops).
  cvt_f32_bf16_kernel<<<dim3(nX / 2048), 256, 0, stream>>>(x, xb);
  cvt_f32_bf16_kernel<<<dim3(nW / 2048), 256, 0, stream>>>(Wq, Wqb);
  cvt_f32_bf16_kernel<<<dim3(nW / 2048), 256, 0, stream>>>(Wk, Wkb);
  cvt_f32_bf16_kernel<<<dim3(nW / 2048), 256, 0, stream>>>(Wv, Wvb);
  cvt_f32_bf16_kernel<<<dim3(nW / 2048), 256, 0, stream>>>(Wo, Wob);

  // 1) QKV projections: 32x64 tile/wave, grid (BT/32/8, E/64, 3).
  dim3 g1(Bc * Tc / 32 / 8, Ec / 64, 3);
  qkv_proj_kernel<<<g1, 256, 0, stream>>>(xb, Wqb, Wkb, Wvb, bq, bk, bv,
                                          Qws, Kws, Vtws);

  // 2) Flash attention: 32-row q-tile per wave, grid (T/32/8, H, B).
  dim3 g2(Tc / 32 / 8, Hc, Bc);
  flash_attn_kernel<<<g2, 256, 0, stream>>>(Qws, Kws, Vtws, attn);

  // 3) Output projection.
  dim3 g3(Bc * Tc / 32 / 8, Ec / 64);
  out_proj_kernel<<<g3, 256, 0, stream>>>(attn, Wob, bo, out);
}
